// VolatilityLSTM_51410758533588
// MI455X (gfx1250) — matrix operand order repaired and checked by reading, not compiled, not run on hardware
//
#include <hip/hip_runtime.h>
#include <hip/hip_bf16.h>

typedef _Float16 h16v __attribute__((ext_vector_type(16)));
typedef _Float16 h8v  __attribute__((ext_vector_type(8)));
typedef _Float16 h4v  __attribute__((ext_vector_type(4)));
typedef float    f8v  __attribute__((ext_vector_type(8)));

#define T_STEPS 22
#define F_IN    12
#define HID     32
#define ROWS    16              // batch rows per wave
#define WAVES_PER_BLOCK 4
#define XW_HALVES (ROWS * T_STEPS * 16)   // 5632 halves, K padded 12->16
#define HB_HALVES (ROWS * HID)            // 512 halves
#define WAVE_HALVES (XW_HALVES + HB_HALVES) // 6144

#define LOG2E 1.4426950408889634f
#define LN2   0.6931471805599453f

__device__ __forceinline__ f8v wmma_f16(h16v a, h16v b, f8v c) {
  return __builtin_amdgcn_wmma_f32_16x16x32_f16(false, a, false, b, (short)0, c, false, false);
}

__device__ __forceinline__ void lds_fence() {
  asm volatile("s_wait_dscnt 0x0" ::: "memory");
}

// Raw hardware transcendentals: v_exp_f32 is 2^x and saturates cleanly at
// +/-inf, so no range guards are needed for sigmoid/tanh.
__device__ __forceinline__ float sigm_f(float x) {
  return __builtin_amdgcn_rcpf(1.0f + __builtin_amdgcn_exp2f(-LOG2E * x));
}
__device__ __forceinline__ float tanh_f(float x) {
  // 1 - 2/(2^{2x*log2e}+1); exp2(+inf)->inf => rcp->0 => 1; exp2(-inf)->0 => -1
  return 1.0f - 2.0f * __builtin_amdgcn_rcpf(1.0f + __builtin_amdgcn_exp2f(2.0f * LOG2E * x));
}

__global__ __launch_bounds__(128) void VolatilityLSTM_51410758533588_kernel(
    const float* __restrict__ x,      // [B, T, F]
    const float* __restrict__ W_ih,   // [128, 12]
    const float* __restrict__ W_hh,   // [128, 32]
    const float* __restrict__ b_ih,   // [128]
    const float* __restrict__ b_hh,   // [128]
    const float* __restrict__ W_head, // [1, 32]
    const float* __restrict__ b_head, // [1]
    float* __restrict__ out,          // [B]
    int B)
{
  __shared__ __align__(16) _Float16 smem[WAVES_PER_BLOCK * WAVE_HALVES];

  const int tid  = threadIdx.x;
  const int warp = tid >> 5;
  const int l    = tid & 31;
  const int n    = l & 15;            // N-col / row index within 16
  const bool hiHalf = (l >= 16);

  const int wid  = blockIdx.x * WAVES_PER_BLOCK + warp;
  const int row0 = wid * ROWS;
  if (row0 >= B) return;              // whole-wave uniform exit

  _Float16* xw = &smem[warp * WAVE_HALVES];
  _Float16* hb = xw + XW_HALVES;

  // ---- Stage x tile -> LDS as f16, K padded to 16 (coalesced b128 reads) ----
  {
    const float* gx = x + (size_t)row0 * (T_STEPS * F_IN);
    #pragma unroll 4
    for (int it = 0; it < 44; ++it) {          // 16 rows * 22 t * 4 chunks / 32 lanes
      int item = it * 32 + l;
      int r   = item / 88;
      int rem = item % 88;
      int t   = rem >> 2;
      int ch  = rem & 3;
      int lidx = (r * T_STEPS + t) * 16 + ch * 4;
      if (ch < 3) {
        const float4 v = *(const float4*)(gx + r * (T_STEPS * F_IN) + t * F_IN + ch * 4);
        h4v hv = { (_Float16)v.x, (_Float16)v.y, (_Float16)v.z, (_Float16)v.w };
        *(h4v*)&xw[lidx] = hv;
      } else {
        h4v z = { (_Float16)0, (_Float16)0, (_Float16)0, (_Float16)0 };
        *(h4v*)&xw[lidx] = z;
      }
    }
    // zero h buffer (h0 = 0)
    h8v z8 = {};
    *(h8v*)&hb[l * 16]     = z8;
    *(h8v*)&hb[l * 16 + 8] = z8;
  }
  lds_fence();

  // ---- Preload WMMA B-operands for W_ih^T (K-padded) and W_hh^T ----
  // B layout (16-bit, 32x16): lane holds col n = l%16; elems 0..7 K=8h..8h+7,
  // elems 8..15 K=16+8h..16+8h+7 (h = l/16), matching the A operand.
  const int kb = hiHalf ? 16 : 0;
  h16v WhhB[8], WihB[8];
  #pragma unroll
  for (int nt = 0; nt < 8; ++nt) {
    const float* wr = W_hh + (nt * 16 + n) * HID + (kb >> 1);
    h16v bh;
    #pragma unroll
    for (int e = 0; e < 16; ++e) bh[e] = (_Float16)wr[(e < 8) ? e : e + 8];
    WhhB[nt] = bh;
    const float* wi = W_ih + (nt * 16 + n) * F_IN;
    h16v bi;
    #pragma unroll
    for (int e = 0; e < 16; ++e) {
      int k = (kb >> 1) + ((e < 8) ? e : e + 8);
      bi[e] = (k < F_IN) ? (_Float16)wi[k] : (_Float16)0;
    }
    WihB[nt] = bi;
  }

  // ---- Per-lane gate biases (b_ih + b_hh), gates i,f,g,o of 32 cols each ----
  const float bi_lo = b_ih[n]      + b_hh[n];
  const float bi_hi = b_ih[n + 16] + b_hh[n + 16];
  const float bf_lo = b_ih[32 + n]      + b_hh[32 + n];
  const float bf_hi = b_ih[32 + n + 16] + b_hh[32 + n + 16];
  const float bg_lo = b_ih[64 + n]      + b_hh[64 + n];
  const float bg_hi = b_ih[64 + n + 16] + b_hh[64 + n + 16];
  const float bo_lo = b_ih[96 + n]      + b_hh[96 + n];
  const float bo_hi = b_ih[96 + n + 16] + b_hh[96 + n + 16];

  // ---- Recurrence: state in registers (C/D layout) ----
  f8v c_lo = {}, c_hi = {}, h_lo = {}, h_hi = {};
  const int kba = hiHalf ? 8 : 0;   // A-operand K base for elements 0..7
  const int r0  = hiHalf ? 8 : 0;   // D-layout row base

  for (int t = 0; t < T_STEPS; ++t) {
    // A-operand for x_t: lane row = n; elems 0..7 = K kba..kba+7 (K>=12 padded 0),
    // elems 8..15 = K 16..23 / 24..31 -> all zero.
    h8v xlo = *(h8v*)&xw[(n * T_STEPS + t) * 16 + kba];
    h16v xa;
    #pragma unroll
    for (int e = 0; e < 8; ++e) { xa[e] = xlo[e]; xa[e + 8] = (_Float16)0; }

    // A-operand for h_{t-1}: elems 0..7 = K kba..kba+7, elems 8..15 = +16.
    h8v hlo = *(h8v*)&hb[n * HID + kba];
    h8v hhi = *(h8v*)&hb[n * HID + kba + 16];
    h16v ha;
    #pragma unroll
    for (int e = 0; e < 8; ++e) { ha[e] = hlo[e]; ha[e + 8] = hhi[e]; }

    f8v z = {};
    // hcols 0..15 : gate tiles 0(i),2(f),4(g),6(o)
    f8v ai = wmma_f16(ha, WhhB[0], wmma_f16(xa, WihB[0], z));
    f8v af = wmma_f16(ha, WhhB[2], wmma_f16(xa, WihB[2], z));
    f8v ag = wmma_f16(ha, WhhB[4], wmma_f16(xa, WihB[4], z));
    f8v ao = wmma_f16(ha, WhhB[6], wmma_f16(xa, WihB[6], z));
    #pragma unroll
    for (int v = 0; v < 8; ++v) {
      float iv = sigm_f(ai[v] + bi_lo);
      float fv = sigm_f(af[v] + bf_lo);
      float gv = tanh_f(ag[v] + bg_lo);
      float ov = sigm_f(ao[v] + bo_lo);
      float cv = fv * c_lo[v] + iv * gv;
      c_lo[v] = cv;
      h_lo[v] = ov * tanh_f(cv);
    }
    // hcols 16..31 : gate tiles 1,3,5,7
    ai = wmma_f16(ha, WhhB[1], wmma_f16(xa, WihB[1], z));
    af = wmma_f16(ha, WhhB[3], wmma_f16(xa, WihB[3], z));
    ag = wmma_f16(ha, WhhB[5], wmma_f16(xa, WihB[5], z));
    ao = wmma_f16(ha, WhhB[7], wmma_f16(xa, WihB[7], z));
    #pragma unroll
    for (int v = 0; v < 8; ++v) {
      float iv = sigm_f(ai[v] + bi_hi);
      float fv = sigm_f(af[v] + bf_hi);
      float gv = tanh_f(ag[v] + bg_hi);
      float ov = sigm_f(ao[v] + bo_hi);
      float cv = fv * c_hi[v] + iv * gv;
      c_hi[v] = cv;
      h_hi[v] = ov * tanh_f(cv);
    }
    // D-layout -> LDS (f16) for next step's A-operand
    #pragma unroll
    for (int v = 0; v < 8; ++v) {
      hb[(v + r0) * HID + n]      = (_Float16)h_lo[v];
      hb[(v + r0) * HID + n + 16] = (_Float16)h_hi[v];
    }
    lds_fence();
  }

  // ---- Head: y = softplus(h @ W_head^T + b_head), reduce via LDS (reuse xw) ----
  float* ho = (float*)xw;  // 16x32 f32 = 2KB, fits in x region
  #pragma unroll
  for (int v = 0; v < 8; ++v) {
    ho[(v + r0) * HID + n]      = h_lo[v];
    ho[(v + r0) * HID + n + 16] = h_hi[v];
  }
  lds_fence();

  if (l < 16) {
    float acc = b_head[0];
    #pragma unroll
    for (int hc = 0; hc < HID; ++hc)
      acc += ho[l * HID + hc] * W_head[hc];
    // softplus(y) = max(y,0) + ln(1 + 2^(-|y|*log2e)); log arg in (1,2]
    float e  = __builtin_amdgcn_exp2f(-LOG2E * fabsf(acc));
    float sp = fmaxf(acc, 0.0f) + LN2 * __builtin_amdgcn_logf(1.0f + e);
    out[row0 + l] = sp;
  }
}

extern "C" void kernel_launch(void* const* d_in, const int* in_sizes, int n_in,
                              void* d_out, int out_size, void* d_ws, size_t ws_size,
                              hipStream_t stream) {
  const float* x      = (const float*)d_in[0];
  const float* W_ih   = (const float*)d_in[1];
  const float* W_hh   = (const float*)d_in[2];
  const float* b_ih   = (const float*)d_in[3];
  const float* b_hh   = (const float*)d_in[4];
  const float* W_head = (const float*)d_in[5];
  const float* b_head = (const float*)d_in[6];
  float* out = (float*)d_out;

  int B = in_sizes[0] / (T_STEPS * F_IN);      // 131072
  int waves  = (B + ROWS - 1) / ROWS;          // 8192
  int blocks = (waves + WAVES_PER_BLOCK - 1) / WAVES_PER_BLOCK;  // 2048

  VolatilityLSTM_51410758533588_kernel<<<blocks, WAVES_PER_BLOCK * 32, 0, stream>>>(
      x, W_ih, W_hh, b_ih, b_hh, W_head, b_head, out, B);
}
